// MultiQuerySelfAttention_61813169324766
// MI455X (gfx1250) — matrix operand order repaired and checked by reading, not compiled, not run on hardware
//
#include <hip/hip_runtime.h>
#include <hip/hip_bf16.h>

typedef __attribute__((ext_vector_type(16))) __bf16       v16bf;
typedef __attribute__((ext_vector_type(8)))  float        v8f;
typedef __attribute__((ext_vector_type(8)))  unsigned int v8ui;

#define BATCH  4
#define SEQ    2048
#define DIM_IN 1024
#define DK     128
#define NQ     8

// 16-bit A-matrix 16x32 per-lane K pattern (contiguous, matches B loads):
// element e -> K = 16*half + e
__device__ __forceinline__ int a_kpat(int e, int half) {
    return (half << 4) + e;
}

__device__ __forceinline__ v8f wmma_bf16(v16bf a, v16bf b, v8f c) {
    return __builtin_amdgcn_wmma_f32_16x16x32_bf16(false, a, false, b, (short)0, c, false, false);
}

// A-fragment loader: 8x dword loads (clang merges into 2x b128: K {0..7} and {8..15} + 16*half)
__device__ __forceinline__ v16bf load_a_frag(const __bf16* __restrict__ base, int half) {
    union { v16bf v; v8ui u; } r;
#pragma unroll
    for (int p = 0; p < 8; ++p)
        r.u[p] = *(const unsigned int*)(base + a_kpat(2 * p, half));
    return r.v;
}

// ---------------------------------------------------------------- fp32 -> bf16
__global__ void cvt_f32_bf16(const float* __restrict__ in, __bf16* __restrict__ out, int n) {
    int i = blockIdx.x * blockDim.x + threadIdx.x;
    int stride = gridDim.x * blockDim.x;
    for (; i < n; i += stride) out[i] = (__bf16)in[i];
}

// ---------------------------- fp32 -> bf16 with transpose: in [w][R][C] -> out [w][C][R]
__global__ void cvt_tr_f32_bf16(const float* __restrict__ in, __bf16* __restrict__ out,
                                int rows, int cols, int batch) {
    size_t rc = (size_t)rows * cols;
    size_t total = rc * batch;
    size_t stride = (size_t)gridDim.x * blockDim.x;
    for (size_t i = blockIdx.x * (size_t)blockDim.x + threadIdx.x; i < total; i += stride) {
        size_t w = i / rc, rem = i % rc;
        int c = (int)(rem / rows), r = (int)(rem % rows);   // contiguous in r -> coalesced writes
        out[i] = (__bf16)in[(w * rows + r) * (size_t)cols + c];
    }
}

// ---------------- projection GEMM: Out[b,w] = X[b] * Wt[w]^T  (Wt is pre-transposed [nw][DK][DIM_IN])
// X: [BATCH][SEQ][DIM_IN] bf16. Each wave computes a 64x16 output tile (4 WMMAs share one B-frag).
// trans==0: Out [BATCH][nw][SEQ][DK] ; trans==1: Out [BATCH][nw][DK][SEQ]
__global__ void proj_gemm(const __bf16* __restrict__ X, const __bf16* __restrict__ Wt,
                          __bf16* __restrict__ Out, int nw, int trans) {
    const int lane = threadIdx.x & 31;
    const int wave = threadIdx.x >> 5;
    const int half = lane >> 4;
    const int nN   = lane & 15;
    const int n0   = (blockIdx.x * 4 + wave) * 64;   // 64-row macro tile
    const int c0   = blockIdx.y * 16;
    const int z = blockIdx.z;
    const int b = z / nw, w = z % nw;

    const __bf16* Xb = X  + (size_t)b * SEQ * DIM_IN;
    const __bf16* Wp = Wt + ((size_t)w * DK + c0 + nN) * DIM_IN;

    v8f acc[4];
#pragma unroll
    for (int i = 0; i < 4; ++i) acc[i] = (v8f){};

    for (int kk = 0; kk < DIM_IN; kk += 32) {
        v16bf bm = *(const v16bf*)(Wp + kk + half * 16);       // 32B vector load
#pragma unroll
        for (int i = 0; i < 4; ++i) {
            v16bf a = load_a_frag(Xb + (size_t)(n0 + i * 16 + nN) * DIM_IN + kk, half);
            acc[i] = wmma_bf16(a, bm, acc[i]);
        }
    }
    if (!trans) {
        __bf16* Ob = Out + (((size_t)b * nw + w) * SEQ) * DK;
#pragma unroll
        for (int i = 0; i < 4; ++i)
#pragma unroll
            for (int r = 0; r < 8; ++r)
                Ob[(size_t)(n0 + i * 16 + r + half * 8) * DK + c0 + nN] = (__bf16)acc[i][r];
    } else {
        __bf16* Ob = Out + (((size_t)b * nw + w) * DK + c0 + nN) * SEQ;
#pragma unroll
        for (int i = 0; i < 4; ++i)
#pragma unroll
            for (int r = 0; r < 8; ++r)                        // consecutive addresses in r
                Ob[n0 + i * 16 + r + half * 8] = (__bf16)acc[i][r];
    }
}

// ------------------------------------------------ fused flash attention (no dk scale)
// Q: [b*h][SEQ][DK], K: [b][SEQ][DK], Vt: [b][DK][SEQ], O: [b*h][SEQ][DK]   (all bf16)
__global__ void mqa_flash(const __bf16* __restrict__ Q, const __bf16* __restrict__ Km,
                          const __bf16* __restrict__ Vt, __bf16* __restrict__ O) {
    __shared__ __bf16 plds[4][16][32];

    const int lane = threadIdx.x & 31;
    const int wave = threadIdx.x >> 5;
    const int half = lane >> 4;
    const int nN   = lane & 15;

    const int bh = blockIdx.y;                 // 0..31
    const int b  = bh >> 3;
    const int n0 = (blockIdx.x * 4 + wave) * 16;

    const __bf16* Qp = Q  + ((size_t)bh * SEQ + n0) * DK;
    const __bf16* Kp = Km + (size_t)b * SEQ * DK;
    const __bf16* Vp = Vt + (size_t)b * DK * SEQ;

    // Preload Q in WMMA A-layout: 4 chunks of K=32 covering dk=128
    v16bf qf[4];
#pragma unroll
    for (int c = 0; c < 4; ++c)
        qf[c] = load_a_frag(Qp + (size_t)nN * DK + c * 32, half);

    v8f acc[8];
#pragma unroll
    for (int t = 0; t < 8; ++t) acc[t] = (v8f){};
    float rmax[8], rsum[8];
#pragma unroll
    for (int r = 0; r < 8; ++r) { rmax[r] = -3.0e38f; rsum[r] = 0.0f; }

    for (int m0 = 0; m0 < SEQ; m0 += 32) {
        // Prefetch next m-chunk of K and V into WGP caches (global_prefetch_b8);
        // uniform guard -> EXEC stays all-ones for the WMMAs below.
        if (m0 + 32 < SEQ) {
            __builtin_prefetch(Kp + (size_t)(m0 + 32 + nN * 2) * DK, 0, 3);
            __builtin_prefetch(Vp + (size_t)(lane * 4) * SEQ + m0 + 32, 0, 3);
        }
        // S = Q * K^T for two 16-wide m tiles; B-frags are 32B vector loads of K rows
        v8f s0 = {}, s1 = {};
#pragma unroll
        for (int c = 0; c < 4; ++c) {
            v16bf b0 = *(const v16bf*)(Kp + (size_t)(m0 + nN) * DK + c * 32 + half * 16);
            v16bf b1 = *(const v16bf*)(Kp + (size_t)(m0 + 16 + nN) * DK + c * 32 + half * 16);
            s0 = wmma_bf16(qf[c], b0, s0);
            s1 = wmma_bf16(qf[c], b1, s1);
        }
        // Online softmax. C-layout: row M=r+8*half lives in one VGPR across a 16-lane half.
        float p0[8], p1[8];
#pragma unroll
        for (int r = 0; r < 8; ++r) {
            float mx = fmaxf(s0[r], s1[r]);
#pragma unroll
            for (int sh = 1; sh < 16; sh <<= 1)
                mx = fmaxf(mx, __shfl_xor(mx, sh, 32));
            float mnew  = fmaxf(rmax[r], mx);
            float alpha = __expf(rmax[r] - mnew);
            rmax[r] = mnew;
            p0[r] = __expf(s0[r] - mnew);
            p1[r] = __expf(s1[r] - mnew);
            float ps = p0[r] + p1[r];
#pragma unroll
            for (int sh = 1; sh < 16; sh <<= 1)
                ps += __shfl_xor(ps, sh, 32);
            rsum[r] = rsum[r] * alpha + ps;
#pragma unroll
            for (int t = 0; t < 8; ++t) acc[t][r] *= alpha;
        }
        // Re-lay out P from C-layout to A-layout through LDS (wave-private tile)
#pragma unroll
        for (int r = 0; r < 8; ++r) {
            int M = r + half * 8;
            plds[wave][M][nN]      = (__bf16)p0[r];
            plds[wave][M][16 + nN] = (__bf16)p1[r];
        }
        asm volatile("s_wait_dscnt 0" ::: "memory");  // DS is in-order per wave; wait for data
        union { v16bf v; v8ui u; } pf;
#pragma unroll
        for (int p = 0; p < 8; ++p)                   // pairs -> ds_load_b32/b128
            pf.u[p] = *(const unsigned int*)(&plds[wave][nN][a_kpat(2 * p, half)]);
        // O += P * V across all 8 v-column tiles; V^T makes B-frags 32B vector loads
#pragma unroll
        for (int t = 0; t < 8; ++t) {
            v16bf vb = *(const v16bf*)(Vp + (size_t)(t * 16 + nN) * SEQ + m0 + half * 16);
            acc[t] = wmma_bf16(pf.v, vb, acc[t]);
        }
    }
    __bf16* Op = O + ((size_t)bh * SEQ + n0) * DK;
#pragma unroll
    for (int t = 0; t < 8; ++t)
#pragma unroll
        for (int r = 0; r < 8; ++r)
            Op[(size_t)(r + half * 8) * DK + t * 16 + nN] = (__bf16)(acc[t][r] / rsum[r]);
}

// ------------------------------ output projection: Y[b,n,d] = sum_h O[b,h,n,:] . Wo[h,d,:]
// One GEMM with K = NQ*DK = 1024 (chunks never straddle a head). Wave computes 16x64.
__global__ void out_gemm(const __bf16* __restrict__ O, const __bf16* __restrict__ Wo,
                         float* __restrict__ Y) {
    const int lane = threadIdx.x & 31;
    const int wave = threadIdx.x >> 5;
    const int half = lane >> 4;
    const int nN   = lane & 15;
    const int n0 = (blockIdx.x * 4 + wave) * 16;
    const int d0 = blockIdx.y * 64;
    const int b  = blockIdx.z;

    v8f acc[4];
#pragma unroll
    for (int j = 0; j < 4; ++j) acc[j] = (v8f){};

    for (int kk = 0; kk < NQ * DK; kk += 32) {
        const int h  = kk >> 7;
        const int vb = kk & 127;
        const __bf16* Oph = O + (((size_t)b * NQ + h) * SEQ + n0) * DK;
        v16bf a = load_a_frag(Oph + (size_t)nN * DK + vb, half);
#pragma unroll
        for (int j = 0; j < 4; ++j) {
            v16bf bm = *(const v16bf*)(Wo + ((size_t)h * DIM_IN + d0 + j * 16 + nN) * DK
                                       + vb + half * 16);
            acc[j] = wmma_bf16(a, bm, acc[j]);
        }
    }
    float* Yp = Y + ((size_t)b * SEQ + n0) * DIM_IN + d0;
#pragma unroll
    for (int j = 0; j < 4; ++j)
#pragma unroll
        for (int r = 0; r < 8; ++r)
            Yp[(size_t)(r + half * 8) * DIM_IN + j * 16 + nN] = acc[j][r];
}

// --------------------------------------------------------------------- launcher
extern "C" void kernel_launch(void* const* d_in, const int* in_sizes, int n_in,
                              void* d_out, int out_size, void* d_ws, size_t ws_size,
                              hipStream_t stream) {
    const float* x  = (const float*)d_in[0];
    const float* pq = (const float*)d_in[1];
    const float* pk = (const float*)d_in[2];
    const float* pv = (const float*)d_in[3];
    const float* po = (const float*)d_in[4];
    float* Y = (float*)d_out;

    char* ws = (char*)d_ws;
    size_t off = 0;
    auto carve = [&](size_t bytes) -> char* {
        off = (off + 255) & ~(size_t)255;
        char* p = ws + off;
        off += bytes;
        return p;
    };
    const size_t NX  = (size_t)BATCH * SEQ * DIM_IN;
    const size_t NWQ = (size_t)NQ * DIM_IN * DK;
    const size_t NWK = (size_t)DIM_IN * DK;
    const size_t NQV = (size_t)BATCH * NQ * SEQ * DK;
    const size_t NKV = (size_t)BATCH * SEQ * DK;

    __bf16* xb  = (__bf16*)carve(NX  * 2);
    __bf16* wqt = (__bf16*)carve(NWQ * 2);   // [8][DK][DIM_IN]
    __bf16* wkt = (__bf16*)carve(NWK * 2);   // [DK][DIM_IN]
    __bf16* wvt = (__bf16*)carve(NWK * 2);   // [DK][DIM_IN]
    __bf16* wo  = (__bf16*)carve(NWQ * 2);   // [8][DIM_IN][DK] (not transposed)
    __bf16* Qb  = (__bf16*)carve(NQV * 2);   // [b*h][SEQ][DK]
    __bf16* Kb  = (__bf16*)carve(NKV * 2);   // [b][SEQ][DK]
    __bf16* Vtb = (__bf16*)carve(NKV * 2);   // [b][DK][SEQ]
    __bf16* Ob  = (__bf16*)carve(NQV * 2);   // [b*h][SEQ][DK]

    cvt_f32_bf16<<<4096, 256, 0, stream>>>(x, xb, (int)NX);
    cvt_tr_f32_bf16<<<1024, 256, 0, stream>>>(pq, wqt, DIM_IN, DK, NQ);
    cvt_tr_f32_bf16<<<256,  256, 0, stream>>>(pk, wkt, DIM_IN, DK, 1);
    cvt_tr_f32_bf16<<<256,  256, 0, stream>>>(pv, wvt, DIM_IN, DK, 1);
    cvt_f32_bf16<<<1024, 256, 0, stream>>>(po, wo, (int)NWQ);

    // Q = x*Wq (8 heads), K = x*Wk, V = x*Wv (V stored transposed for the PV GEMM)
    proj_gemm<<<dim3(SEQ / 256, DK / 16, BATCH * NQ), 128, 0, stream>>>(xb, wqt, Qb, NQ, 0);
    proj_gemm<<<dim3(SEQ / 256, DK / 16, BATCH),      128, 0, stream>>>(xb, wkt, Kb, 1, 0);
    proj_gemm<<<dim3(SEQ / 256, DK / 16, BATCH),      128, 0, stream>>>(xb, wvt, Vtb, 1, 1);

    // fused softmax(QK^T)V
    mqa_flash<<<dim3(SEQ / 64, BATCH * NQ), 128, 0, stream>>>(Qb, Kb, Vtb, Ob);

    // Y = sum_h O_h Wo_h^T
    out_gemm<<<dim3(SEQ / 64, DIM_IN / 64, BATCH), 128, 0, stream>>>(Ob, wo, Y);
}
